// GraphUNet_28381143892879
// MI455X (gfx1250) — 8-bit matrix operand order repaired and checked by reading, compiled, not run on hardware
//
#include <hip/hip_runtime.h>
#include <hip/hip_bf16.h>
#include <math.h>

typedef float v2f __attribute__((ext_vector_type(2)));
typedef float v8f __attribute__((ext_vector_type(8)));
typedef int   v8i __attribute__((ext_vector_type(8)));
typedef unsigned int u32x4 __attribute__((ext_vector_type(4)));
typedef int   i32x8 __attribute__((ext_vector_type(8)));
typedef int   i32x4 __attribute__((ext_vector_type(4)));

#define NNODE 4096
#define NEDGE 131072
#define INDIM 16
#define HDIM  256


__global__ void zero32_k(unsigned* __restrict__ w, long n) {
  long t = (long)blockIdx.x * blockDim.x + threadIdx.x;
  if (t < n) w[t] = 0u;
}

__global__ void edges_k(const int* __restrict__ ei, unsigned char* __restrict__ Au, int n, int e) {
  int t = blockIdx.x * blockDim.x + threadIdx.x;
  if (t >= e) return;
  int src = ei[t];
  int dst = ei[e + t];
  Au[(size_t)dst * n + src] = 1;
}

__global__ void diag_k(unsigned char* __restrict__ Au, int n) {
  int i = blockIdx.x * blockDim.x + threadIdx.x;
  if (i < n) Au[(size_t)i * n + i] = 1;
}

__global__ void deg_k(const unsigned char* __restrict__ Au, float* __restrict__ deg, int n) {
  int i = blockIdx.x * blockDim.x + threadIdx.x;
  if (i >= n) return;
  const unsigned* w = (const unsigned*)(Au + (size_t)i * n);
  int s = 0;
  for (int j = 0; j < (n >> 2); ++j) s += __popc(w[j]);
  deg[i] = (float)s;
}

__global__ void gemm_f32_k(const float* __restrict__ A, const float* __restrict__ B,
                           float* __restrict__ C, int M, int K, int N) {
  const unsigned lane = threadIdx.x & 31u, wave = threadIdx.x >> 5u;
  const int mrow = blockIdx.y * 16;
  const int ncol = blockIdx.x * 64 + (int)wave * 16;
  if (ncol >= N || mrow >= M) return;
  const int ar = mrow + (int)(lane & 15u);
  const int kh = (int)(lane >> 4u) * 2;
  const int bc = ncol + (int)(lane & 15u);
  v8f acc = {};
  for (int k = 0; k < K; k += 4) {
    v2f a, b;
    a.x = A[(size_t)ar * K + k + kh];
    a.y = A[(size_t)ar * K + k + kh + 1];
    b.x = B[(size_t)(k + kh) * N + bc];
    b.y = B[(size_t)(k + kh + 1) * N + bc];
    acc = __builtin_amdgcn_wmma_f32_16x16x4_f32(false, a, false, b, (short)0, acc,
                                                false, false);
  }
  const int rbase = mrow + (int)(lane >> 4u) * 8;
#pragma unroll
  for (int r = 0; r < 8; ++r)
    C[(size_t)(rbase + r) * N + bc] = acc[r];
}

__global__ void gemm_aggr_k(const unsigned char* __restrict__ Au, const float* __restrict__ B,
                            float* __restrict__ C, int M, int K, int N,
                            const float* __restrict__ deg, const float* __restrict__ bias,
                            int relu) {
  __shared__ float Bs[64 * 64];
  const unsigned lane = threadIdx.x & 31u, wave = threadIdx.x >> 5u;
  const int mrow = blockIdx.y * 16;
  const int nblk = blockIdx.x * 64;
  const int ar = mrow + (int)(lane & 15u);
  const int kh = (int)(lane >> 4u) * 2;
  const int cc = (int)wave * 16 + (int)(lane & 15u);
  const int bc = nblk + cc;
  const unsigned ldsoff = (unsigned)(size_t)&Bs[0];
  v8f acc = {};

  for (int k = 0; k < K; k += 64) {
    __syncthreads();
    if (wave == 0) {
      unsigned long long ga = (unsigned long long)(size_t)(B + (size_t)k * N + nblk);
      u32x4 g0;
      g0[0] = 1u;
      g0[1] = ldsoff;
      g0[2] = (unsigned)(ga & 0xFFFFFFFFull);
      g0[3] = (unsigned)((ga >> 32) & 0x1FFFFFFull)
              | (2u << 30);
      i32x8 g1;
      const unsigned d0 = 64u, d1 = 64u;
      const unsigned long long s0 = (unsigned long long)N;
      g1[0] = (int)(2u << 16);
      g1[1] = (int)((d0 & 0xFFFFu) << 16);
      g1[2] = (int)((d0 >> 16) | ((d1 & 0xFFFFu) << 16));
      g1[3] = (int)((d1 >> 16) | (64u << 16));
      g1[4] = (int)64u;
      g1[5] = (int)(s0 & 0xFFFFFFFFull);
      g1[6] = (int)((s0 >> 32) & 0xFFFFull);
      g1[7] = 0;
      i32x4 gz4 = {0, 0, 0, 0};
      i32x8 gz8 = {0, 0, 0, 0, 0, 0, 0, 0};
      __builtin_amdgcn_tensor_load_to_lds(g0, g1, gz4, gz4, gz8, 0);
      __builtin_amdgcn_s_wait_tensorcnt(0);
    }
    __syncthreads();
#pragma unroll
    for (int kk = 0; kk < 64; kk += 4) {
      v2f a, b;
      a.x = (float)Au[(size_t)ar * K + k + kk + kh];
      a.y = (float)Au[(size_t)ar * K + k + kk + kh + 1];
      b.x = Bs[(kk + kh) * 64 + cc];
      b.y = Bs[(kk + kh + 1) * 64 + cc];
      acc = __builtin_amdgcn_wmma_f32_16x16x4_f32(false, a, false, b, (short)0, acc,
                                                  false, false);
    }
  }
  const int rbase = mrow + (int)(lane >> 4u) * 8;
#pragma unroll
  for (int r = 0; r < 8; ++r) {
    int row = rbase + r;
    float v = acc[r] / deg[row] + bias[bc];
    if (relu) v = fmaxf(v, 0.0f);
    C[(size_t)row * N + bc] = v;
  }
}

__global__ void gemm_mask_k(const unsigned char* __restrict__ Rows,
                            const unsigned char* __restrict__ ColsT,
                            unsigned char* __restrict__ AuOut, int kdim, int nc) {
  const unsigned lane = threadIdx.x & 31u, wave = threadIdx.x >> 5u;
  const int mrow = blockIdx.y * 16;
  const int ncol = blockIdx.x * 64 + (int)wave * 16;
  if (ncol >= kdim || mrow >= kdim) return;
  const int am = mrow + (int)(lane & 15u);
  const int bn = ncol + (int)(lane & 15u);
  const int hf = (int)(lane >> 4u);
  v8i acc = {};
  for (int k = 0; k < nc; k += 64) {
    v8i a, b;
#pragma unroll
    for (int v = 0; v < 8; ++v) {
      int ka = k + (v >> 1) * 16 + (v & 1) * 4 + hf * 8;
      a[v] = *(const int*)(Rows + (size_t)am * nc + ka);
      int kb = k + (v >> 1) * 16 + (v & 1) * 4 + hf * 8;
      b[v] = *(const int*)(ColsT + (size_t)bn * nc + kb);
    }
    acc = __builtin_amdgcn_wmma_i32_16x16x64_iu8(false, a, false, b, acc, false, false);
  }
  const int rbase = mrow + hf * 8;
#pragma unroll
  for (int r = 0; r < 8; ++r) {
    int row = rbase + r;
    AuOut[(size_t)row * kdim + bn] = (row == bn) ? (unsigned char)1
                                                 : (unsigned char)(acc[r] != 0);
  }
}


__global__ void score_k(const float* __restrict__ h, const float* __restrict__ p,
                        float* __restrict__ score, int n, int H) {
  int i = blockIdx.x * blockDim.x + threadIdx.x;
  if (i >= n) return;
  float pn = 0.f, s = 0.f;
  for (int c = 0; c < H; ++c) { pn += p[c] * p[c]; s += h[(size_t)i * H + c] * p[c]; }
  score[i] = tanhf(s * rsqrtf(pn));
}

__global__ void topk_k(const float* __restrict__ score, int* __restrict__ perm,
                       float* __restrict__ vals, int n, int kkeep) {
  __shared__ float ss[4096];
  __shared__ int   si[4096];
  const int tid = threadIdx.x;
  for (int i = tid; i < n; i += blockDim.x) { ss[i] = score[i]; si[i] = i; }
  __syncthreads();
  for (int k = 2; k <= n; k <<= 1) {
    for (int j = k >> 1; j > 0; j >>= 1) {
      for (int i = tid; i < n; i += blockDim.x) {
        int x = i ^ j;
        if (x > i) {
          bool desc = ((i & k) == 0);
          float a = ss[i], b = ss[x];
          int ia = si[i], ib = si[x];
          bool aFirst = (a > b) || (a == b && ia < ib);
          if (desc ? !aFirst : aFirst) { ss[i] = b; ss[x] = a; si[i] = ib; si[x] = ia; }
        }
      }
      __syncthreads();
    }
  }
  for (int i = tid; i < kkeep; i += blockDim.x) { perm[i] = si[i]; vals[i] = ss[i]; }
}

__global__ void gatherh_k(const float* __restrict__ h, const int* __restrict__ perm,
                          const float* __restrict__ vals, float* __restrict__ out,
                          int k, int H) {
  long t = (long)blockIdx.x * blockDim.x + threadIdx.x;
  if (t >= (long)k * H) return;
  int a = (int)(t / H), c = (int)(t % H);
  out[t] = h[(size_t)perm[a] * H + c] * vals[a];
}

__global__ void gatherrows_k(const unsigned char* __restrict__ Au, const int* __restrict__ perm,
                             unsigned char* __restrict__ out, int k, int nc) {
  long t = (long)blockIdx.x * blockDim.x + threadIdx.x;
  if (t >= (long)k * nc) return;
  int a = (int)(t / nc), c = (int)(t % nc);
  out[t] = Au[(size_t)perm[a] * nc + c];
}

__global__ void gathercolsT_k(const unsigned char* __restrict__ Au, const int* __restrict__ perm,
                              unsigned char* __restrict__ out, int k, int nc) {
  long t = (long)blockIdx.x * blockDim.x + threadIdx.x;
  if (t >= (long)k * nc) return;
  int a = (int)(t / nc), c = (int)(t % nc);
  out[t] = Au[(size_t)c * nc + perm[a]];
}

__global__ void copyf_k(float* __restrict__ o, const float* __restrict__ in, long n) {
  long t = (long)blockIdx.x * blockDim.x + threadIdx.x;
  if (t < n) o[t] = in[t];
}

__global__ void scatteradd_k(float* __restrict__ out, const float* __restrict__ h,
                             const int* __restrict__ perm, int k, int H) {
  long t = (long)blockIdx.x * blockDim.x + threadIdx.x;
  if (t >= (long)k * H) return;
  int a = (int)(t / H), c = (int)(t % H);
  out[(size_t)perm[a] * H + c] += h[t];
}


__global__ void dotw_k(const float* __restrict__ h, const float* __restrict__ w,
                       float* __restrict__ z, int n, int H) {
  int i = blockIdx.x * blockDim.x + threadIdx.x;
  if (i >= n) return;
  float s = 0.f;
  for (int c = 0; c < H; ++c) s += h[(size_t)i * H + c] * w[c];
  z[i] = s;
}

__global__ void logits_k(const unsigned char* __restrict__ Au, const float* __restrict__ z,
                         const float* __restrict__ deg, const float* __restrict__ bout,
                         float* __restrict__ logits, int n) {
  int i = blockIdx.x * blockDim.x + threadIdx.x;
  if (i >= n) return;
  float s = 0.f;
  const unsigned char* row = Au + (size_t)i * n;
  for (int j = 0; j < n; ++j) s += (float)row[j] * z[j];
  logits[i] = s / deg[i] + bout[0];
}

__global__ void loss_k(const float* __restrict__ logits, const float* __restrict__ labels,
                       float* __restrict__ out, int n) {
  __shared__ float red[1024];
  const int tid = threadIdx.x;
  float ls = 0.f;
  for (int i = tid; i < n; i += blockDim.x) ls += labels[i];
  red[tid] = ls; __syncthreads();
  for (int s = blockDim.x / 2; s > 0; s >>= 1) { if (tid < s) red[tid] += red[tid + s]; __syncthreads(); }
  float frac = red[0] / (float)n;
  __syncthreads();
  float acc = 0.f;
  for (int i = tid; i < n; i += blockDim.x) {
    float x = logits[i], y = labels[i];
    out[1 + i] = 1.f / (1.f + expf(-x));
    float w = y * (1.f - frac) + (1.f - y) * frac;
    float lsp = fminf(x, 0.f) - log1pf(expf(-fabsf(x)));
    float lsn = fminf(-x, 0.f) - log1pf(expf(-fabsf(x)));
    acc += w * (-(y * lsp + (1.f - y) * lsn));
  }
  red[tid] = acc; __syncthreads();
  for (int s = blockDim.x / 2; s > 0; s >>= 1) { if (tid < s) red[tid] += red[tid + s]; __syncthreads(); }
  if (tid == 0) out[0] = red[0] / (float)n;
}


static inline unsigned cdiv(long a, long b) { return (unsigned)((a + b - 1) / b); }

extern "C" void kernel_launch(void* const* d_in, const int* in_sizes, int n_in,
                              void* d_out, int out_size, void* d_ws, size_t ws_size,
                              hipStream_t stream) {
  (void)in_sizes; (void)n_in; (void)out_size; (void)ws_size;
  const float* x     = (const float*)d_in[0];
  const int*   ei    = (const int*)d_in[1];
  const float* labels= (const float*)d_in[2];
  const float* Wd0   = (const float*)d_in[3];
  const float* bd0   = (const float*)d_in[4];
  const float* Wd    = (const float*)d_in[5];
  const float* bd    = (const float*)d_in[6];
  const float* Wu    = (const float*)d_in[7];
  const float* bu    = (const float*)d_in[8];
  const float* Wout  = (const float*)d_in[9];
  const float* bout  = (const float*)d_in[10];
  const float* P     = (const float*)d_in[11];

  const int n0 = NNODE, H = HDIM;
  const int nl[4] = { 4096, 2048, 1024, 512 };

  char* p = (char*)d_ws;
  auto carve = [&](size_t bytes) -> char* {
    char* r = p; p += (bytes + 255) & ~(size_t)255; return r;
  };
  unsigned char* AU[4];
  for (int l = 0; l < 4; ++l) AU[l] = (unsigned char*)carve((size_t)nl[l] * nl[l]);
  unsigned char* ROWS = (unsigned char*)carve((size_t)2048 * 4096);
  unsigned char* COLT = (unsigned char*)carve((size_t)2048 * 4096);
  float* DEG[4];
  for (int l = 0; l < 4; ++l) DEG[l] = (float*)carve((size_t)nl[l] * 4);
  int* PERM[3];
  for (int l = 0; l < 3; ++l) PERM[l] = (int*)carve((size_t)nl[l + 1] * 4);
  float* VALS  = (float*)carve((size_t)2048 * 4);
  float* SCORE = (float*)carve((size_t)4096 * 4);
  float* XS0 = (float*)carve((size_t)4096 * H * 4);
  float* XS1 = (float*)carve((size_t)2048 * H * 4);
  float* XS2 = (float*)carve((size_t)1024 * H * 4);
  float* HA  = (float*)carve((size_t)4096 * H * 4);
  float* HB  = (float*)carve((size_t)4096 * H * 4);
  float* XW  = (float*)carve((size_t)4096 * H * 4);
  float* Z   = (float*)carve((size_t)4096 * 4);
  float* LG  = (float*)carve((size_t)4096 * 4);

  zero32_k<<<cdiv((long)n0 * n0 / 4, 256), 256, 0, stream>>>((unsigned*)AU[0], (long)n0 * n0 / 4);
  edges_k<<<cdiv(NEDGE, 256), 256, 0, stream>>>(ei, AU[0], n0, NEDGE);
  diag_k<<<cdiv(n0, 256), 256, 0, stream>>>(AU[0], n0);
  deg_k<<<cdiv(n0, 256), 256, 0, stream>>>(AU[0], DEG[0], n0);

  gemm_f32_k<<<dim3(H / 64, n0 / 16), 128, 0, stream>>>(x, Wd0, XW, n0, INDIM, H);
  gemm_aggr_k<<<dim3(H / 64, n0 / 16), 128, 0, stream>>>(AU[0], XW, XS0, n0, n0, H,
                                                         DEG[0], bd0, 1);

  for (int lvl = 0; lvl < 3; ++lvl) {
    const int nc = nl[lvl], kk = nl[lvl + 1];
    const float* hcur = (lvl == 0) ? XS0 : (lvl == 1) ? XS1 : XS2;
    float*       hout = (lvl == 0) ? XS1 : (lvl == 1) ? XS2 : HA;

    score_k<<<cdiv(nc, 256), 256, 0, stream>>>(hcur, P + (size_t)lvl * H, SCORE, nc, H);
    topk_k<<<1, 1024, 0, stream>>>(SCORE, PERM[lvl], VALS, nc, kk);
    gatherh_k<<<cdiv((long)kk * H, 256), 256, 0, stream>>>(hcur, PERM[lvl], VALS, HB, kk, H);

    gatherrows_k<<<cdiv((long)kk * nc, 256), 256, 0, stream>>>(AU[lvl], PERM[lvl], ROWS, kk, nc);
    gathercolsT_k<<<cdiv((long)kk * nc, 256), 256, 0, stream>>>(AU[lvl], PERM[lvl], COLT, kk, nc);
    gemm_mask_k<<<dim3(kk / 64, kk / 16), 128, 0, stream>>>(ROWS, COLT, AU[lvl + 1], kk, nc);
    deg_k<<<cdiv(kk, 256), 256, 0, stream>>>(AU[lvl + 1], DEG[lvl + 1], kk);

    gemm_f32_k<<<dim3(H / 64, kk / 16), 128, 0, stream>>>(HB, Wd + (size_t)lvl * H * H, XW, kk, H, H);
    gemm_aggr_k<<<dim3(H / 64, kk / 16), 128, 0, stream>>>(AU[lvl + 1], XW, hout, kk, kk, H,
                                                           DEG[lvl + 1], bd + (size_t)lvl * H, 1);
  }

  copyf_k<<<cdiv((long)1024 * H, 256), 256, 0, stream>>>(HB, XS2, (long)1024 * H);
  scatteradd_k<<<cdiv((long)512 * H, 256), 256, 0, stream>>>(HB, HA, PERM[2], 512, H);
  gemm_f32_k<<<dim3(H / 64, 1024 / 16), 128, 0, stream>>>(HB, Wu, XW, 1024, H, H);
  gemm_aggr_k<<<dim3(H / 64, 1024 / 16), 128, 0, stream>>>(AU[2], XW, HA, 1024, 1024, H,
                                                           DEG[2], bu, 1);
  copyf_k<<<cdiv((long)2048 * H, 256), 256, 0, stream>>>(HB, XS1, (long)2048 * H);
  scatteradd_k<<<cdiv((long)1024 * H, 256), 256, 0, stream>>>(HB, HA, PERM[1], 1024, H);
  gemm_f32_k<<<dim3(H / 64, 2048 / 16), 128, 0, stream>>>(HB, Wu + (size_t)H * H, XW, 2048, H, H);
  gemm_aggr_k<<<dim3(H / 64, 2048 / 16), 128, 0, stream>>>(AU[1], XW, HA, 2048, 2048, H,
                                                           DEG[1], bu + H, 1);
  copyf_k<<<cdiv((long)4096 * H, 256), 256, 0, stream>>>(HB, XS0, (long)4096 * H);
  scatteradd_k<<<cdiv((long)2048 * H, 256), 256, 0, stream>>>(HB, HA, PERM[0], 2048, H);
  dotw_k<<<cdiv(4096, 256), 256, 0, stream>>>(HB, Wout, Z, 4096, H);
  logits_k<<<cdiv(4096, 256), 256, 0, stream>>>(AU[0], Z, DEG[0], bout, LG, 4096);

  loss_k<<<1, 1024, 0, stream>>>(LG, labels, (float*)d_out, 4096);
}
